// KANTextGenerator_81320910782611
// MI455X (gfx1250) — matrix operand order repaired and checked by reading, not compiled, not run on hardware
//
#include <hip/hip_runtime.h>
#include <hip/hip_bf16.h>

typedef __bf16 bf16_t;
typedef __attribute__((ext_vector_type(16))) __bf16 v16bf;
typedef __attribute__((ext_vector_type(8)))  __bf16 v8bf;
typedef __attribute__((ext_vector_type(4)))  __bf16 v4bf;
typedef __attribute__((ext_vector_type(8)))  float  v8f;

#define KSTEP 64
#define LDSW  72   // 64 + 8 bf16 pad -> 144B row pitch, conflict-free b128 column reads

// async global -> LDS, 16 bytes per lane, GVS addressing (saddr64 + vgpr offset)
#define ASYNC_LDS_B128(ldsaddr, goff, base)                              \
  asm volatile("global_load_async_to_lds_b128 %0, %1, %2"                \
               :: "v"(ldsaddr), "v"(goff), "s"(base) : "memory")

// ---------------- embedding gather + f32 -> bf16 ----------------
__global__ __launch_bounds__(256) void embed_kernel(
    const int* __restrict__ ids, const float* __restrict__ emb,
    bf16_t* __restrict__ x, int D)
{
  const int tok = blockIdx.x;
  const int id  = ids[tok];
  const float4 v = ((const float4*)(emb + (size_t)id * D))[threadIdx.x];
  v4bf o;
  o[0] = (bf16_t)v.x; o[1] = (bf16_t)v.y; o[2] = (bf16_t)v.z; o[3] = (bf16_t)v.w;
  *(v4bf*)(x + (size_t)tok * D + threadIdx.x * 4) = o;
}

// ---------------- LDS-tiled transpose -> bf16 (out[c][r] = in[r][c]) ----------------
template<typename T>
__global__ __launch_bounds__(256) void transpose_to_bf16_kernel(
    const T* __restrict__ in, bf16_t* __restrict__ out,
    int R, int C, size_t inBatchStride, size_t outBatchStride)
{
  __shared__ bf16_t tile[32][34];
  in  += (size_t)blockIdx.z * inBatchStride;
  out += (size_t)blockIdx.z * outBatchStride;
  const int c = blockIdx.x * 32 + threadIdx.x;
  #pragma unroll
  for (int i = 0; i < 4; ++i) {
    const int r = blockIdx.y * 32 + threadIdx.y + i * 8;
    tile[threadIdx.y + i * 8][threadIdx.x] = (bf16_t)in[(size_t)r * C + c];
  }
  __syncthreads();
  const int rT = blockIdx.y * 32 + threadIdx.x;
  #pragma unroll
  for (int i = 0; i < 4; ++i) {
    const int cT = blockIdx.x * 32 + threadIdx.y + i * 8;
    out[(size_t)cT * R + rT] = tile[threadIdx.x][threadIdx.y + i * 8];
  }
}

// ---------------- row softmax f32 -> bf16 ----------------
__global__ __launch_bounds__(256) void softmax_kernel(
    const float* __restrict__ scores, bf16_t* __restrict__ weights, int Ncol)
{
  __shared__ float red[8];
  const size_t row = blockIdx.x;
  const float* s = scores + row * (size_t)Ncol;
  bf16_t* w = weights + row * (size_t)Ncol;
  const int lane = threadIdx.x & 31;
  const int wave = threadIdx.x >> 5;
  const int per  = Ncol >> 8;           // cols / 256 (== 8 here)

  float v[16];
  float m = -3.402823466e38f;
  for (int i = 0; i < per; ++i) {
    v[i] = s[threadIdx.x + (i << 8)];
    m = fmaxf(m, v[i]);
  }
  #pragma unroll
  for (int o = 16; o >= 1; o >>= 1) m = fmaxf(m, __shfl_xor(m, o, 32));
  if (lane == 0) red[wave] = m;
  __syncthreads();
  m = red[0];
  #pragma unroll
  for (int j = 1; j < 8; ++j) m = fmaxf(m, red[j]);
  __syncthreads();

  float sum = 0.f;
  for (int i = 0; i < per; ++i) { v[i] = __expf(v[i] - m); sum += v[i]; }
  #pragma unroll
  for (int o = 16; o >= 1; o >>= 1) sum += __shfl_xor(sum, o, 32);
  if (lane == 0) red[wave] = sum;
  __syncthreads();
  sum = 0.f;
  #pragma unroll
  for (int j = 0; j < 8; ++j) sum += red[j];
  const float inv = 1.0f / sum;
  for (int i = 0; i < per; ++i)
    w[threadIdx.x + (i << 8)] = (bf16_t)(v[i] * inv);
}

// ---------------- NT bf16 GEMM: async double-buffered LDS + v_wmma ----------------
// C[M,N] = A[M,K] * Bt[N,K]^T (+bias) (+relu).  M % 256, N % 128, K % 64.
// Macro tile 256x128, 8 waves (4 M-waves x 2 N-waves), wave tile 64x64 = 4x4 WMMA tiles.
// Per 32-K sub-step: 16 ds_load_b128 feed 16 v_wmma (1.0 LDS load per WMMA, 4x frag reuse).
template<bool OUT_BF16, bool RELU, bool HAS_BIAS>
__global__ __launch_bounds__(256) void gemm_bf16_wmma(
    const bf16_t* __restrict__ A, const bf16_t* __restrict__ Bt,
    void* __restrict__ Cout, const float* __restrict__ bias,
    int M, int N, int K,
    size_t strideA, size_t strideB, size_t strideC)
{
  __shared__ bf16_t ldsA[2][256 * LDSW];   // 73,728 B
  __shared__ bf16_t ldsB[2][128 * LDSW];   // 36,864 B  (total ~108 KB of 320 KB/WGP)

  A  += (size_t)blockIdx.z * strideA;
  Bt += (size_t)blockIdx.z * strideB;

  const int rowBase = blockIdx.y * 256;
  const int colBase = blockIdx.x * 128;

  const int tid    = threadIdx.x;
  const int lane   = tid & 31;
  const int waveId = tid >> 5;
  const int wm     = waveId & 3;   // 4 waves over M: 64 rows each
  const int wn     = waveId >> 2;  // 2 waves over N: 64 cols each
  const int laneLo = lane & 15;
  const int laneHi = lane >> 4;

  v8f acc[4][4];
  #pragma unroll
  for (int mt = 0; mt < 4; ++mt)
    #pragma unroll
    for (int nt = 0; nt < 4; ++nt)
      #pragma unroll
      for (int e = 0; e < 8; ++e) acc[mt][nt][e] = 0.f;

  // cooperative tile load mapping:
  //  A tile 256x64: one row per thread  -> 8x b128
  //  B tile 128x64: half row per thread -> 4x b128
  const int ldrB = tid >> 1;
  const int ldcB = (tid & 1) * 32;
  const unsigned long long baseA = (unsigned long long)(A  + (size_t)rowBase * K);
  const unsigned long long baseB = (unsigned long long)(Bt + (size_t)colBase * K);
  const unsigned gOffA = (unsigned)((unsigned)tid * (unsigned)K * 2u);
  const unsigned gOffB = (unsigned)(((unsigned)ldrB * (unsigned)K + (unsigned)ldcB) * 2u);

  unsigned sAaddr[2], sBaddr[2];
  sAaddr[0] = (unsigned)(uintptr_t)&ldsA[0][tid * LDSW];
  sAaddr[1] = (unsigned)(uintptr_t)&ldsA[1][tid * LDSW];
  sBaddr[0] = (unsigned)(uintptr_t)&ldsB[0][ldrB * LDSW + ldcB];
  sBaddr[1] = (unsigned)(uintptr_t)&ldsB[1][ldrB * LDSW + ldcB];

  auto issue_tile = [&](int buf, int kt) {
    const unsigned gA = gOffA + (unsigned)(kt * 2);
    #pragma unroll
    for (int q = 0; q < 8; ++q)
      ASYNC_LDS_B128(sAaddr[buf] + q * 16, gA + q * 16, baseA);
    const unsigned gB = gOffB + (unsigned)(kt * 2);
    #pragma unroll
    for (int q = 0; q < 4; ++q)
      ASYNC_LDS_B128(sBaddr[buf] + q * 16, gB + q * 16, baseB);
  };

  // prologue: tile 0 -> buffer 0 (12 async b128 per thread in flight)
  issue_tile(0, 0);

  int buf = 0;
  for (int kt = 0; kt < K; kt += KSTEP, buf ^= 1) {
    __syncthreads();                       // all waves done reading buf^1
    if (kt + KSTEP < K) {
      issue_tile(buf ^ 1, kt + KSTEP);     // prefetch next tile
      asm volatile("s_wait_asynccnt 12" ::: "memory");  // current tile landed (in-order)
    } else {
      asm volatile("s_wait_asynccnt 0" ::: "memory");
    }
    __syncthreads();                       // everyone's current tile visible

    const bf16_t* bA = ldsA[buf];
    const bf16_t* bB = ldsB[buf];
    #pragma unroll
    for (int ks = 0; ks < KSTEP; ks += 32) {
      // A fragment per ISA 16-bit A layout: lanes 0-15 k(0..7,16..23), lanes 16-31 k(8..15,24..31)
      v16bf fa[4];
      #pragma unroll
      for (int mt = 0; mt < 4; ++mt) {
        const bf16_t* p = bA + (wm * 64 + mt * 16 + laneLo) * LDSW + ks + 8 * laneHi;
        v8bf lo = *(const v8bf*)p;
        v8bf hi = *(const v8bf*)(p + 16);
        #pragma unroll
        for (int e = 0; e < 8; ++e) { fa[mt][e] = lo[e]; fa[mt][e + 8] = hi[e]; }
      }
      // B fragment: row n of Bt, same k order as A: lanes 0-15 k(0..7,16..23), lanes 16-31 k(8..15,24..31)
      v16bf fb[4];
      #pragma unroll
      for (int nt = 0; nt < 4; ++nt) {
        const bf16_t* p = bB + (wn * 64 + nt * 16 + laneLo) * LDSW + ks + 8 * laneHi;
        v8bf lo = *(const v8bf*)p;
        v8bf hi = *(const v8bf*)(p + 16);
        #pragma unroll
        for (int e = 0; e < 8; ++e) { fb[nt][e] = lo[e]; fb[nt][e + 8] = hi[e]; }
      }

      #pragma unroll
      for (int mt = 0; mt < 4; ++mt)
        #pragma unroll
        for (int nt = 0; nt < 4; ++nt)
          acc[mt][nt] = __builtin_amdgcn_wmma_f32_16x16x32_bf16(
              false, fa[mt], false, fb[nt], (short)0, acc[mt][nt], false, false);
    }
  }

  // epilogue: C/D layout -> m = base + r + 8*laneHi, n = base + laneLo
  #pragma unroll
  for (int nt = 0; nt < 4; ++nt) {
    const int n = colBase + wn * 64 + nt * 16 + laneLo;
    const float bv = HAS_BIAS ? bias[n] : 0.f;
    #pragma unroll
    for (int mt = 0; mt < 4; ++mt) {
      const int mBase = rowBase + wm * 64 + mt * 16 + 8 * laneHi;
      #pragma unroll
      for (int r = 0; r < 8; ++r) {
        float v = acc[mt][nt][r] + bv;
        if (RELU) v = fmaxf(v, 0.f);
        const size_t idx = (size_t)blockIdx.z * strideC + (size_t)(mBase + r) * N + n;
        if (OUT_BF16) ((bf16_t*)Cout)[idx] = (bf16_t)v;
        else          ((float*)Cout)[idx]  = v;
      }
    }
  }
}

extern "C" void kernel_launch(void* const* d_in, const int* in_sizes, int n_in,
                              void* d_out, int out_size, void* d_ws, size_t ws_size,
                              hipStream_t stream)
{
  const int*   ids = (const int*)d_in[0];
  const float* emb = (const float*)d_in[1];
  const float* W1  = (const float*)d_in[2];
  const float* b1  = (const float*)d_in[3];
  const float* W2  = (const float*)d_in[4];
  const float* b2  = (const float*)d_in[5];

  constexpr int    Bb = 4, S = 2048, D = 1024, H = 4096, V = 32000;
  constexpr size_t T  = (size_t)Bb * S;   // 8192 tokens

  char* ws = (char*)d_ws;
  size_t off = 0;
  auto carve = [&](size_t bytes) -> char* {
    char* p = ws + off;
    off = (off + bytes + 255) & ~(size_t)255;
    return p;
  };
  bf16_t* x_bf   = (bf16_t*)carve(T * D * 2);                 // [B,S,D]
  bf16_t* xT_bf  = (bf16_t*)carve(T * D * 2);                 // [B,D,S]
  bf16_t* W1t    = (bf16_t*)carve((size_t)H * D * 2);         // [H,D]
  bf16_t* W2t    = (bf16_t*)carve((size_t)V * H * 2);         // [V,H]
  float*  scores = (float*)carve((size_t)Bb * S * S * 4);     // [B,S,S]
  bf16_t* wts    = (bf16_t*)carve((size_t)Bb * S * S * 2);    // [B,S,S]
  bf16_t* att    = (bf16_t*)carve(T * D * 2);                 // [B,S,D]
  bf16_t* hid    = (bf16_t*)carve(T * H * 2);                 // [B*S,H]

  // 1) embedding gather -> bf16
  embed_kernel<<<dim3((unsigned)T), 256, 0, stream>>>(ids, emb, x_bf, D);

  // 2) transposes (NT layouts for the WMMA GEMMs)
  const dim3 tb(32, 8);
  transpose_to_bf16_kernel<bf16_t><<<dim3(D / 32, S / 32, Bb), tb, 0, stream>>>(
      x_bf, xT_bf, S, D, (size_t)S * D, (size_t)S * D);
  transpose_to_bf16_kernel<float><<<dim3(H / 32, D / 32, 1), tb, 0, stream>>>(
      W1, W1t, D, H, 0, 0);
  transpose_to_bf16_kernel<float><<<dim3(V / 32, H / 32, 1), tb, 0, stream>>>(
      W2, W2t, H, V, 0, 0);

  // 3) scores[b] = x[b] @ x[b]^T   (A = x, Bt = x)
  gemm_bf16_wmma<false, false, false><<<dim3(S / 128, S / 256, Bb), 256, 0, stream>>>(
      x_bf, x_bf, scores, nullptr, S, S, D,
      (size_t)S * D, (size_t)S * D, (size_t)S * S);

  // 4) softmax rows -> bf16 weights
  softmax_kernel<<<dim3((unsigned)T), 256, 0, stream>>>(scores, wts, S);

  // 5) attended[b] = weights[b] @ x[b]   (Bt = xT[b] : [D,S])
  gemm_bf16_wmma<true, false, false><<<dim3(D / 128, S / 256, Bb), 256, 0, stream>>>(
      wts, xT_bf, att, nullptr, S, D, S,
      (size_t)S * S, (size_t)D * S, (size_t)S * D);

  // 6) hidden = relu(att @ W1 + b1)   (Bt = W1t : [H,D])
  gemm_bf16_wmma<true, true, true><<<dim3(H / 128, (unsigned)(T / 256), 1), 256, 0, stream>>>(
      att, W1t, hid, b1, (int)T, H, D, 0, 0, 0);

  // 7) logits = hidden @ W2 + b2 -> f32 d_out   (Bt = W2t : [V,H])
  gemm_bf16_wmma<false, false, true><<<dim3(V / 128, (unsigned)(T / 256), 1), 256, 0, stream>>>(
      hid, W2t, d_out, b2, (int)T, V, H, 0, 0, 0);
}
